// CaptionNetWithAttantion2LSTM_79156247265658
// MI455X (gfx1250) — matrix operand order repaired and checked by reading, not compiled, not run on hardware
//
#include <hip/hip_runtime.h>

// ---------------- problem dims ----------------
constexpr int Bn = 128;    // batch
constexpr int Tn = 24;     // time steps
constexpr int Vn = 12000;  // vocab
constexpr int En = 512;    // embed dim
constexpr int Hn = 1024;   // hidden
constexpr int Fn = 2048;   // image feature dim
constexpr int G4 = 4 * Hn; // 4096 gate width
constexpr int EF = En + Fn;

// ---------------- WMMA fragment types ----------------
typedef _Float16 half8  __attribute__((ext_vector_type(8)));
typedef _Float16 half16 __attribute__((ext_vector_type(16)));
typedef float    fl8    __attribute__((ext_vector_type(8)));

// A-matrix 16x32 f16 fragment (wave32):
// lanes 0-15 (hiH=0): K = k0+0..15 ; lanes 16-31 (hiH=1): K = k0+16..31, 16 contiguous halves
__device__ __forceinline__ half16 frag_a(const _Float16* p /* row + k0 + hiH*16 */) {
    half8 lo = *(const half8*)p;
    half8 hi = *(const half8*)(p + 8);
    half16 r;
#pragma unroll
    for (int i = 0; i < 8; ++i) { r[i] = lo[i]; r[i + 8] = hi[i]; }
    return r;
}

// B-matrix 32x16 f16 fragment (wave32), B[k,n] = W[n,k] (row-major W):
// lanes 0-15 hold K=k0+0..15, lanes 16-31 hold K=k0+16..31, 16 contiguous halves
__device__ __forceinline__ half16 frag_b(const _Float16* p /* Wrow + k0 + hiH*16 */) {
    half8 lo = *(const half8*)p;
    half8 hi = *(const half8*)(p + 8);
    half16 r;
#pragma unroll
    for (int i = 0; i < 8; ++i) { r[i] = lo[i]; r[i + 8] = hi[i]; }
    return r;
}

#define WMMA_F16(A, B, C) \
    __builtin_amdgcn_wmma_f32_16x16x32_f16(false, (A), false, (B), (short)0, (C), false, false)

__device__ __forceinline__ float sigf(float x) { return 1.0f / (1.0f + __expf(-x)); }
__device__ __forceinline__ float tanh_fast(float x) {
    float e = __expf(-2.0f * fabsf(x));
    float t = (1.0f - e) / (1.0f + e);
    return x < 0.0f ? -t : t;
}

// Flat pointer -> wave-relative LDS byte offset (flat LDS aperture keeps the
// offset in addr[31:0]; hardware adds LDS_BASE for both flat and async paths).
__device__ __forceinline__ unsigned lds_off32(const void* p) {
    return (unsigned)(unsigned long long)p;
}

// CDNA5 async copy: 16B global -> LDS per lane, tracked by ASYNCcnt.
__device__ __forceinline__ void async_copy_b128(unsigned lds_byte_off, const _Float16* g) {
    asm volatile("global_load_async_to_lds_b128 %0, %1, off"
                 :: "v"(lds_byte_off), "v"(g)
                 : "memory");
}
__device__ __forceinline__ void wait_async0() {
    asm volatile("s_wait_asynccnt 0x0" ::: "memory");
}

// ---------------- setup kernels ----------------
__global__ void cvt_f16(const float* __restrict__ s, _Float16* __restrict__ d, long n) {
    long i = (long)blockIdx.x * blockDim.x + threadIdx.x;
    long st = (long)gridDim.x * blockDim.x;
    for (; i < n; i += st) d[i] = (_Float16)s[i];
}

__global__ void addcvt_f16(const float* __restrict__ a, const float* __restrict__ b,
                           _Float16* __restrict__ d, long n) {
    long i = (long)blockIdx.x * blockDim.x + threadIdx.x;
    long st = (long)gridDim.x * blockDim.x;
    for (; i < n; i += st) d[i] = (_Float16)(a[i] + b[i]);
}

__global__ void add_f32(const float* __restrict__ a, const float* __restrict__ b,
                        float* __restrict__ d, int n) {
    int i = blockIdx.x * blockDim.x + threadIdx.x;
    if (i < n) d[i] = a[i] + b[i];
}

// W_ihE = W_ih1[:, 0:E] converted to f16 ([4H, E])
__global__ void slice_cvt(const float* __restrict__ s /*[4H,EF]*/, _Float16* __restrict__ d /*[4H,E]*/) {
    int i = blockIdx.x * blockDim.x + threadIdx.x; // 4H*E
    int j = i >> 9, e = i & 511;
    d[i] = (_Float16)s[(size_t)j * EF + e];
}

// cap16[(b*T+t), e] = f16(emb[ix[b,t], e])
__global__ void gather_emb(const float* __restrict__ emb, const int* __restrict__ ix,
                           _Float16* __restrict__ d) {
    int i = blockIdx.x * blockDim.x + threadIdx.x; // B*T*E
    int r = i >> 9, e = i & 511;
    d[i] = (_Float16)emb[(size_t)ix[r] * En + e];
}

// h0 = c0 = img @ W_img.T + b_img ; write c (f32) and h (f16)
__global__ void init_h0(const float* __restrict__ img, const float* __restrict__ Wimg,
                        const float* __restrict__ bimg, float* __restrict__ c,
                        _Float16* __restrict__ h16d) {
    int idx = blockIdx.x * blockDim.x + threadIdx.x; // B*H
    int b = idx >> 10, j = idx & 1023;
    const float* ir = img + (size_t)b * Fn;
    const float* wr = Wimg + (size_t)j * Fn;
    float s = bimg[j];
    for (int f = 0; f < Fn; ++f) s = fmaf(ir[f], wr[f], s);
    c[idx] = s;
    h16d[idx] = (_Float16)s;
}

// img_gate[b, j4] = img @ W_ih1[:, E:].T + b_ih1 + b_hh1  (time-invariant lstm1 gate part)
__global__ void init_imggate(const float* __restrict__ img, const float* __restrict__ Wih1,
                             const float* __restrict__ bih1, const float* __restrict__ bhh1,
                             float* __restrict__ ig) {
    int idx = blockIdx.x * blockDim.x + threadIdx.x; // B*4H
    int b = idx >> 12, j = idx & 4095;
    const float* ir = img + (size_t)b * Fn;
    const float* wr = Wih1 + (size_t)j * EF + En;
    float s = bih1[j] + bhh1[j];
    for (int f = 0; f < Fn; ++f) s = fmaf(ir[f], wr[f], s);
    ig[idx] = s;
}

// ---------------- recurrent step kernels ----------------
// lstm1: gates = h @ W_hh1.T + emb_t @ W_ihE.T + img_gate  -> cell(c) -> h1 (f16), c1 (f32)
// grid(64) x block(256): wave w -> m-tile (batch rows w*16..), blockIdx.x -> n-tile within H
__global__ __launch_bounds__(256) void lstm1_step(
    const _Float16* __restrict__ hp,   const _Float16* __restrict__ Whh,
    const _Float16* __restrict__ cap,  const _Float16* __restrict__ WihE,
    const float* __restrict__ imggate, const float* __restrict__ cin,
    float* __restrict__ cout, _Float16* __restrict__ h1out, int t) {
    const int lane = threadIdx.x & 31;
    const int wv   = threadIdx.x >> 5;
    const int m0   = wv * 16;
    const int n0   = blockIdx.x * 16;
    const int lo16 = lane & 15;
    const int hiH  = lane >> 4;
    const int mA   = m0 + lo16;
    const int nB   = n0 + lo16;

    fl8 acc0 = {}, acc1 = {}, acc2 = {}, acc3 = {};

    const _Float16* aRow = hp + (size_t)mA * Hn;
    const _Float16* b0R  = Whh + (size_t)(nB         ) * Hn;
    const _Float16* b1R  = Whh + (size_t)(nB +     Hn) * Hn;
    const _Float16* b2R  = Whh + (size_t)(nB + 2 * Hn) * Hn;
    const _Float16* b3R  = Whh + (size_t)(nB + 3 * Hn) * Hn;
    for (int k0 = 0; k0 < Hn; k0 += 32) {
        half16 a  = frag_a(aRow + k0 + hiH * 16);
        half16 b0 = frag_b(b0R + k0 + hiH * 16);
        half16 b1 = frag_b(b1R + k0 + hiH * 16);
        half16 b2 = frag_b(b2R + k0 + hiH * 16);
        half16 b3 = frag_b(b3R + k0 + hiH * 16);
        acc0 = WMMA_F16(a, b0, acc0);
        acc1 = WMMA_F16(a, b1, acc1);
        acc2 = WMMA_F16(a, b2, acc2);
        acc3 = WMMA_F16(a, b3, acc3);
    }
    const _Float16* eRow = cap + ((size_t)mA * Tn + t) * En;
    const _Float16* e0R  = WihE + (size_t)(nB         ) * En;
    const _Float16* e1R  = WihE + (size_t)(nB +     Hn) * En;
    const _Float16* e2R  = WihE + (size_t)(nB + 2 * Hn) * En;
    const _Float16* e3R  = WihE + (size_t)(nB + 3 * Hn) * En;
    for (int k0 = 0; k0 < En; k0 += 32) {
        half16 a  = frag_a(eRow + k0 + hiH * 16);
        half16 b0 = frag_b(e0R + k0 + hiH * 16);
        half16 b1 = frag_b(e1R + k0 + hiH * 16);
        half16 b2 = frag_b(e2R + k0 + hiH * 16);
        half16 b3 = frag_b(e3R + k0 + hiH * 16);
        acc0 = WMMA_F16(a, b0, acc0);
        acc1 = WMMA_F16(a, b1, acc1);
        acc2 = WMMA_F16(a, b2, acc2);
        acc3 = WMMA_F16(a, b3, acc3);
    }
    // epilogue: D VGPR r -> M = hiH*8 + r, N = lane&15
#pragma unroll
    for (int r = 0; r < 8; ++r) {
        int m = m0 + hiH * 8 + r;
        int n = n0 + lo16;
        size_t gi = (size_t)m * G4 + n;
        float ig = acc0[r] + imggate[gi];
        float fg = acc1[r] + imggate[gi + Hn];
        float gg = acc2[r] + imggate[gi + 2 * Hn];
        float og = acc3[r] + imggate[gi + 3 * Hn];
        size_t hidx = (size_t)m * Hn + n;
        float cn = sigf(fg) * cin[hidx] + sigf(ig) * tanh_fast(gg);
        float hn = sigf(og) * tanh_fast(cn);
        cout[hidx]  = cn;
        h1out[hidx] = (_Float16)hn;
    }
}

// lstm2: x = h = h1, so gates = h1 @ (W_ih2+W_hh2).T + (b_ih2+b_hh2) -> cell(c1)
// outputs: c (f32, next step), h (f16, next step), h2_all[(b*T+t), :] (f16)
__global__ __launch_bounds__(256) void lstm2_step(
    const _Float16* __restrict__ h1p, const _Float16* __restrict__ W2,
    const float* __restrict__ b2, const float* __restrict__ cin,
    float* __restrict__ cout, _Float16* __restrict__ hout,
    _Float16* __restrict__ h2all, int t) {
    const int lane = threadIdx.x & 31;
    const int wv   = threadIdx.x >> 5;
    const int m0   = wv * 16;
    const int n0   = blockIdx.x * 16;
    const int lo16 = lane & 15;
    const int hiH  = lane >> 4;
    const int mA   = m0 + lo16;
    const int nB   = n0 + lo16;

    fl8 acc0 = {}, acc1 = {}, acc2 = {}, acc3 = {};
    const _Float16* aRow = h1p + (size_t)mA * Hn;
    const _Float16* b0R  = W2 + (size_t)(nB         ) * Hn;
    const _Float16* b1R  = W2 + (size_t)(nB +     Hn) * Hn;
    const _Float16* b2R  = W2 + (size_t)(nB + 2 * Hn) * Hn;
    const _Float16* b3R  = W2 + (size_t)(nB + 3 * Hn) * Hn;
    for (int k0 = 0; k0 < Hn; k0 += 32) {
        half16 a  = frag_a(aRow + k0 + hiH * 16);
        half16 b0 = frag_b(b0R + k0 + hiH * 16);
        half16 b1 = frag_b(b1R + k0 + hiH * 16);
        half16 b2 = frag_b(b2R + k0 + hiH * 16);
        half16 b3 = frag_b(b3R + k0 + hiH * 16);
        acc0 = WMMA_F16(a, b0, acc0);
        acc1 = WMMA_F16(a, b1, acc1);
        acc2 = WMMA_F16(a, b2, acc2);
        acc3 = WMMA_F16(a, b3, acc3);
    }
#pragma unroll
    for (int r = 0; r < 8; ++r) {
        int m = m0 + hiH * 8 + r;
        int n = n0 + lo16;
        float ig = acc0[r] + b2[n];
        float fg = acc1[r] + b2[n + Hn];
        float gg = acc2[r] + b2[n + 2 * Hn];
        float og = acc3[r] + b2[n + 3 * Hn];
        size_t hidx = (size_t)m * Hn + n;
        float cn = sigf(fg) * cin[hidx] + sigf(ig) * tanh_fast(gg);
        float hn = sigf(og) * tanh_fast(cn);
        cout[hidx] = cn;
        _Float16 hh = (_Float16)hn;
        hout[hidx] = hh;
        h2all[((size_t)m * Tn + t) * Hn + n] = hh;
    }
}

// ---------------- final projection ----------------
// out = h2_all @ W_out.T + b_out : [3072,1024] x [12000,1024]^T -> d_out [B,T,V]
// Block: 8 waves stacked along M (128 rows); each wave computes 16 x 160 (10 n-tiles)
// to amortize A fragments. B chunk (160 x 64 f16) staged in LDS once per block via
// CDNA5 async global->LDS copies and shared by all 8 waves (ds_load fragments).
constexpr int NTW   = 10;        // n-tiles per wave
constexpr int BLK_N = 16 * NTW;  // 160 columns per block
constexpr int KC    = 64;        // K-chunk (two WMMA k-steps)
constexpr int KP    = KC + 8;    // padded LDS row stride (halves) -> conflict-free

__global__ __launch_bounds__(256) void out_gemm(
    const _Float16* __restrict__ h2all, const _Float16* __restrict__ Wout,
    const float* __restrict__ bout, float* __restrict__ out) {
    __shared__ __align__(16) _Float16 ldsB[BLK_N * KP]; // 160*72*2 = 23040 B

    const int lane = threadIdx.x & 31;
    const int wv   = threadIdx.x >> 5;
    const int tid  = threadIdx.x;
    const int m0   = blockIdx.y * 128 + wv * 16;
    const int nblk = blockIdx.x * BLK_N;
    const int lo16 = lane & 15;
    const int hiH  = lane >> 4;

    fl8 acc[NTW];
#pragma unroll
    for (int i = 0; i < NTW; ++i) acc[i] = (fl8){};

    const _Float16* aRow = h2all + (size_t)(m0 + lo16) * Hn;

    for (int kc = 0; kc < Hn; kc += KC) {
        // cooperative async stage of B chunk: 160 rows x 64 halves = 1280 x b128
#pragma unroll
        for (int q = 0; q < (BLK_N * KC / 8) / 256; ++q) { // 5 iterations
            int l   = q * 256 + tid;
            int row = l >> 3;        // 8 x b128 per row
            int c8  = (l & 7) * 8;   // half offset within chunk
            async_copy_b128(lds_off32(&ldsB[row * KP + c8]),
                            Wout + (size_t)(nblk + row) * Hn + kc + c8);
        }
        wait_async0();
        __syncthreads();

#pragma unroll
        for (int sub = 0; sub < 2; ++sub) {
            half16 a = frag_a(aRow + kc + sub * 32 + hiH * 16);
            const _Float16* bBase = &ldsB[sub * 32 + hiH * 16];
#pragma unroll
            for (int nt = 0; nt < NTW; ++nt) {
                half16 b = frag_b(bBase + (nt * 16 + lo16) * KP);
                acc[nt] = WMMA_F16(a, b, acc[nt]);
            }
        }
        __syncthreads(); // protect LDS before next chunk overwrites
    }

#pragma unroll
    for (int nt = 0; nt < NTW; ++nt) {
#pragma unroll
        for (int r = 0; r < 8; ++r) {
            int m = m0 + hiH * 8 + r;
            int n = nblk + nt * 16 + lo16;
            out[(size_t)m * Vn + n] = acc[nt][r] + bout[n];
        }
    }
}

// ---------------- workspace layout (bytes, 256-aligned) ----------------
constexpr size_t OFF_WHH1    = 0;                          // 4H*H f16   = 8388608
constexpr size_t OFF_WSUM2   = OFF_WHH1   + 8388608;       // 4H*H f16
constexpr size_t OFF_WOUT    = OFF_WSUM2  + 8388608;       // V*H f16    = 24576000
constexpr size_t OFF_WIHE    = OFF_WOUT   + 24576000;      // 4H*E f16   = 4194304
constexpr size_t OFF_CAP     = OFF_WIHE   + 4194304;       // B*T*E f16  = 3145728
constexpr size_t OFF_IMGGATE = OFF_CAP    + 3145728;       // B*4H f32   = 2097152
constexpr size_t OFF_B2      = OFF_IMGGATE+ 2097152;       // 4H f32     = 16384
constexpr size_t OFF_H16     = OFF_B2     + 16384;         // B*H f16    = 262144
constexpr size_t OFF_H116    = OFF_H16    + 262144;        // B*H f16
constexpr size_t OFF_C       = OFF_H116   + 262144;        // B*H f32    = 524288
constexpr size_t OFF_C1      = OFF_C      + 524288;        // B*H f32
constexpr size_t OFF_H2ALL   = OFF_C1     + 524288;        // B*T*H f16  = 6291456
// total ~58.7 MB

extern "C" void kernel_launch(void* const* d_in, const int* in_sizes, int n_in,
                              void* d_out, int out_size, void* d_ws, size_t ws_size,
                              hipStream_t stream) {
    (void)in_sizes; (void)n_in; (void)out_size; (void)ws_size;
    const float* img   = (const float*)d_in[0];
    const int*   capix = (const int*)d_in[1];
    const float* Wimg  = (const float*)d_in[2];
    const float* bimg  = (const float*)d_in[3];
    const float* emb   = (const float*)d_in[4];
    // d_in[5..10]: Wa, ba, Ua, ub, va, vb — dead code in the reference (softmax over a
    // size-1 axis makes weight==1, so context == image_vectors); intentionally unused.
    const float* Wih1  = (const float*)d_in[11];
    const float* Whh1  = (const float*)d_in[12];
    const float* bih1  = (const float*)d_in[13];
    const float* bhh1  = (const float*)d_in[14];
    const float* Wih2  = (const float*)d_in[15];
    const float* Whh2  = (const float*)d_in[16];
    const float* bih2  = (const float*)d_in[17];
    const float* bhh2  = (const float*)d_in[18];
    const float* Wout  = (const float*)d_in[19];
    const float* bout  = (const float*)d_in[20];

    char* ws = (char*)d_ws;
    _Float16* wWhh1  = (_Float16*)(ws + OFF_WHH1);
    _Float16* wWsum2 = (_Float16*)(ws + OFF_WSUM2);
    _Float16* wWout  = (_Float16*)(ws + OFF_WOUT);
    _Float16* wWihE  = (_Float16*)(ws + OFF_WIHE);
    _Float16* wCap   = (_Float16*)(ws + OFF_CAP);
    float*    wIG    = (float*)   (ws + OFF_IMGGATE);
    float*    wB2    = (float*)   (ws + OFF_B2);
    _Float16* wH16   = (_Float16*)(ws + OFF_H16);
    _Float16* wH116  = (_Float16*)(ws + OFF_H116);
    float*    wC     = (float*)   (ws + OFF_C);
    float*    wC1    = (float*)   (ws + OFF_C1);
    _Float16* wH2all = (_Float16*)(ws + OFF_H2ALL);

    // one-time setup (all on stream; fully rewrites every ws region it reads later)
    cvt_f16   <<<1024, 256, 0, stream>>>(Whh1, wWhh1, (long)G4 * Hn);
    addcvt_f16<<<1024, 256, 0, stream>>>(Wih2, Whh2, wWsum2, (long)G4 * Hn);
    cvt_f16   <<<2048, 256, 0, stream>>>(Wout, wWout, (long)Vn * Hn);
    slice_cvt <<<(G4 * En) / 256, 256, 0, stream>>>(Wih1, wWihE);
    add_f32   <<<16, 256, 0, stream>>>(bih2, bhh2, wB2, G4);
    gather_emb<<<(Bn * Tn * En) / 256, 256, 0, stream>>>(emb, capix, wCap);
    init_h0   <<<(Bn * Hn) / 256, 256, 0, stream>>>(img, Wimg, bimg, wC, wH16);
    init_imggate<<<(Bn * G4) / 256, 256, 0, stream>>>(img, Wih1, bih1, bhh1, wIG);

    // sequential recurrence: 24 steps, two fused GEMM+cell kernels each
    for (int t = 0; t < Tn; ++t) {
        lstm1_step<<<Hn / 16, 256, 0, stream>>>(wH16, wWhh1, wCap, wWihE, wIG, wC, wC1, wH116, t);
        lstm2_step<<<Hn / 16, 256, 0, stream>>>(wH116, wWsum2, wB2, wC1, wC, wH16, wH2all, t);
    }

    // final projection: [3072,1024] x [12000,1024]^T + b_out -> d_out
    out_gemm<<<dim3(Vn / BLK_N, (Bn * Tn) / 128), 256, 0, stream>>>(wH2all, wWout, bout, (float*)d_out);
}
